// MSCrossAttnBlock_17008070492338
// MI455X (gfx1250) — compile-verified, matrix operand order checked by reading, not run on hardware
//
#include <hip/hip_runtime.h>
#include <cstdint>
#include <cstddef>

// Problem constants (match reference)
#define BB   4
#define EE   32
#define LQn  1024      // E*E
#define DDim 1024
#define NHn  16
#define NPn  4
#define DHn  64
#define KKn  1024

typedef __attribute__((ext_vector_type(16))) __bf16 v16bf;
typedef __attribute__((ext_vector_type(8)))  __bf16 v8bf;
typedef __attribute__((ext_vector_type(4)))  __bf16 v4bf;
typedef __attribute__((ext_vector_type(8)))  float  v8f;

#if defined(__HIP_DEVICE_COMPILE__) && __has_builtin(__builtin_amdgcn_tensor_load_to_lds) && __has_builtin(__builtin_amdgcn_s_wait_tensorcnt)
#define HAVE_TDM 1
#else
#define HAVE_TDM 0
#endif

#if HAVE_TDM
typedef __attribute__((ext_vector_type(4))) unsigned int u32x4;
typedef __attribute__((ext_vector_type(8))) int          i32x8;
typedef __attribute__((ext_vector_type(4))) int          i32x4;

// Issue a TDM DMA of a (64 rows x 128 cols) bf16 tile of W (row stride K=1024
// elements) into LDS at byte offset lds_off. 2D tile -> groups 2/3 (and the
// extended group) zero. D# packing per CDNA5 ISA ch.8 (group0:
// count/lds/global/type, group1: data_size, dims, tile dims, stride).
// This toolchain exposes the 6-arg builtin:
//   (u32x4 g0, i32x8 g1, i32x4 g2, i32x4 g3, i32x8 gx, int cpol)
__device__ __forceinline__ void tdm_load_tile(unsigned lds_off,
                                              const __bf16* gsrc, int Nrows) {
  unsigned long long ga = (unsigned long long)(uintptr_t)gsrc;
  u32x4 g0;
  g0[0] = 1u;                                            // count=1, user mode
  g0[1] = lds_off;                                       // lds_addr (bytes)
  g0[2] = (unsigned)ga;                                  // global_addr[31:0]
  g0[3] = (unsigned)((ga >> 32) & 0x1FFFFFFu) | (0x2u << 30);  // addr hi | type=2
  i32x8 g1;
  g1[0] = 0x00010000;                                    // data_size=1 (2 bytes)
  g1[1] = (int)((KKn & 0xFFFFu) << 16);                  // tensor_dim0 lo16
  g1[2] = (int)(((unsigned)Nrows & 0xFFFFu) << 16);      // dim0 hi16=0 | dim1 lo16
  g1[3] = (int)((128u << 16) | (((unsigned)Nrows >> 16) & 0xFFFFu)); // tile_dim0=128
  g1[4] = 64;                                            // tile_dim1=64, tile_dim2=0
  g1[5] = KKn;                                           // tensor_dim0_stride lo32
  g1[6] = 0;
  g1[7] = 0;
  i32x4 z4 = {0, 0, 0, 0};
  i32x8 z8 = {0, 0, 0, 0, 0, 0, 0, 0};
  __builtin_amdgcn_tensor_load_to_lds(g0, g1, z4, z4, z8, 0);
}
#endif

__device__ __forceinline__ __bf16 f2bf(float f) {
  unsigned int u = __float_as_uint(f);
  u += 0x7fffu + ((u >> 16) & 1u);           // round-to-nearest-even
  unsigned short h = (unsigned short)(u >> 16);
  __bf16 r;
  __builtin_memcpy(&r, &h, sizeof(r));
  return r;
}

// ---------------------------------------------------------------------------
// fp32 -> bf16 conversion (vectorized by 4; nelem % 4 == 0 for all callers)
// ---------------------------------------------------------------------------
__global__ __launch_bounds__(256) void cvt_bf16_kernel(
    const float* __restrict__ x, __bf16* __restrict__ y, size_t nvec4) {
  size_t i = blockIdx.x * (size_t)blockDim.x + threadIdx.x;
  size_t stride = (size_t)gridDim.x * blockDim.x;
  for (; i < nvec4; i += stride) {
    float4 v = reinterpret_cast<const float4*>(x)[i];
    v4bf o;
    o[0] = f2bf(v.x); o[1] = f2bf(v.y); o[2] = f2bf(v.z); o[3] = f2bf(v.w);
    reinterpret_cast<v4bf*>(y)[i] = o;
  }
}

// ---------------------------------------------------------------------------
// LayerNorm over last dim (1024), output bf16.
// Output row = n*out_bstride_rows + q  (lets feat segments interleave).
// ---------------------------------------------------------------------------
__global__ __launch_bounds__(256) void ln_bf16_kernel(
    const float* __restrict__ x, const float* __restrict__ g,
    const float* __restrict__ b, __bf16* __restrict__ y,
    int rows_per_batch, long long out_bstride_rows) {
  __shared__ float s_sum[256];
  __shared__ float s_sq[256];
  const int r = blockIdx.x;
  const int n = r / rows_per_batch;
  const int q = r % rows_per_batch;
  const float* xr = x + (size_t)r * DDim;
  __bf16* yr = y + ((size_t)n * out_bstride_rows + q) * DDim;
  const int tid = threadIdx.x;
  const int c = tid * 4;
  float4 v = *reinterpret_cast<const float4*>(xr + c);
  s_sum[tid] = v.x + v.y + v.z + v.w;
  s_sq[tid] = v.x * v.x + v.y * v.y + v.z * v.z + v.w * v.w;
  __syncthreads();
  for (int st = 128; st > 0; st >>= 1) {
    if (tid < st) { s_sum[tid] += s_sum[tid + st]; s_sq[tid] += s_sq[tid + st]; }
    __syncthreads();
  }
  const float mean = s_sum[0] * (1.0f / DDim);
  const float var = s_sq[0] * (1.0f / DDim) - mean * mean;
  const float inv = rsqrtf(var + 1e-6f);
  float4 gg = *reinterpret_cast<const float4*>(g + c);
  float4 bb = *reinterpret_cast<const float4*>(b + c);
  v4bf o;
  o[0] = f2bf((v.x - mean) * inv * gg.x + bb.x);
  o[1] = f2bf((v.y - mean) * inv * gg.y + bb.y);
  o[2] = f2bf((v.z - mean) * inv * gg.z + bb.z);
  o[3] = f2bf((v.w - mean) * inv * gg.w + bb.w);
  *reinterpret_cast<v4bf*>(yr + c) = o;
}

// ---------------------------------------------------------------------------
// bf16 WMMA GEMM:  C(M,N) = A(M,K) * W(N,K)^T + bias   (K = 1024)
// Block: 256 threads = 8 waves stacked in M. Block tile 256x64, wave 32x64.
// W tile double-buffered in LDS; on CDNA5 the Tensor Data Mover DMAs the
// next 64x128 bf16 chunk (tensor_load_to_lds, TENSORcnt) while the current
// chunk feeds v_wmma_f32_16x16x32_bf16. A is streamed from global (L2-hot).
// ---------------------------------------------------------------------------
__global__ __launch_bounds__(256) void gemm_bf16_wmma(
    const __bf16* __restrict__ A, const __bf16* __restrict__ W,
    const float* __restrict__ bias, float* __restrict__ C, int M, int N) {
  __shared__ unsigned short lds_raw[2][64 * 128];   // 2 x 16 KB double buffer

  const int tid = threadIdx.x;
  const int lane = tid & 31;
  const int wave = tid >> 5;
  const int l16 = lane & 15;
  const int hv = lane >> 4;                      // lane half: 0 or 1
  const int m0 = blockIdx.x * 256 + wave * 32;
  const int n0 = blockIdx.y * 64;

  v8f acc[2][4];
  const v8f zero = {0.f, 0.f, 0.f, 0.f, 0.f, 0.f, 0.f, 0.f};
#pragma unroll
  for (int i = 0; i < 2; ++i)
#pragma unroll
    for (int j = 0; j < 4; ++j) acc[i][j] = zero;

  const __bf16* a_row0 = A + (size_t)(m0 + l16) * KKn;
  const __bf16* a_row1 = a_row0 + (size_t)16 * KKn;
  const __bf16* wtile = W + (size_t)n0 * KKn;

#if HAVE_TDM
  const unsigned lds_off0 = (unsigned)(uintptr_t)(&lds_raw[0][0]);
  const unsigned lds_off1 = (unsigned)(uintptr_t)(&lds_raw[1][0]);
  if (wave == 0) tdm_load_tile(lds_off0, wtile, N);   // prologue DMA chunk 0
#endif

  int buf = 0;
  for (int kc = 0; kc < KKn / 128; ++kc) {
#if HAVE_TDM
    if (wave == 0) {
      if (kc + 1 < KKn / 128) {
        tdm_load_tile(buf ? lds_off0 : lds_off1, wtile + (kc + 1) * 128, N);
        __builtin_amdgcn_s_wait_tensorcnt(1);    // chunk kc landed in LDS
      } else {
        __builtin_amdgcn_s_wait_tensorcnt(0);
      }
    }
    __syncthreads();                             // publish chunk kc to all waves
#else
    // fallback: cooperative load of W[n0..n0+63, kc*128..+127] into LDS
#pragma unroll
    for (int cchunk = 0; cchunk < 4; ++cchunk) {
      int chunk = tid + cchunk * 256;            // 0..1023 b128 chunks
      int row = chunk >> 4;
      int col8 = (chunk & 15) << 3;
      *reinterpret_cast<v8bf*>(
          reinterpret_cast<__bf16*>(&lds_raw[buf][0]) + row * 128 + col8) =
          *reinterpret_cast<const v8bf*>(&wtile[(size_t)row * KKn + kc * 128 + col8]);
    }
    __syncthreads();
#endif

    if (kc + 1 < KKn / 128) {                    // prefetch next A chunk
      __builtin_prefetch(a_row0 + (kc + 1) * 128, 0, 1);
      __builtin_prefetch(a_row1 + (kc + 1) * 128, 0, 1);
    }

    const __bf16* ldsb = reinterpret_cast<const __bf16*>(&lds_raw[buf][0]);
#pragma unroll
    for (int ks = 0; ks < 4; ++ks) {
      const int klo = kc * 128 + ks * 32 + hv * 8;
      v16bf afrag[2], bfrag[4];
      {
        v8bf lo = *reinterpret_cast<const v8bf*>(a_row0 + klo);
        v8bf hi = *reinterpret_cast<const v8bf*>(a_row0 + klo + 16);
        afrag[0] = __builtin_shufflevector(lo, hi, 0, 1, 2, 3, 4, 5, 6, 7, 8, 9,
                                           10, 11, 12, 13, 14, 15);
        lo = *reinterpret_cast<const v8bf*>(a_row1 + klo);
        hi = *reinterpret_cast<const v8bf*>(a_row1 + klo + 16);
        afrag[1] = __builtin_shufflevector(lo, hi, 0, 1, 2, 3, 4, 5, 6, 7, 8, 9,
                                           10, 11, 12, 13, 14, 15);
      }
#pragma unroll
      for (int j = 0; j < 4; ++j) {
        const __bf16* wrow = ldsb + (j * 16 + l16) * 128 + ks * 32 + hv * 8;
        v8bf lo = *reinterpret_cast<const v8bf*>(wrow);
        v8bf hi = *reinterpret_cast<const v8bf*>(wrow + 16);
        bfrag[j] = __builtin_shufflevector(lo, hi, 0, 1, 2, 3, 4, 5, 6, 7, 8, 9,
                                           10, 11, 12, 13, 14, 15);
      }
#pragma unroll
      for (int i = 0; i < 2; ++i)
#pragma unroll
        for (int j = 0; j < 4; ++j)
          acc[i][j] = __builtin_amdgcn_wmma_f32_16x16x32_bf16(
              false, afrag[i], false, bfrag[j], (short)0, acc[i][j], false, false);
    }
    __syncthreads();                             // reads of buf done -> reusable
    buf ^= 1;
  }

  // epilogue: bias + store (C/D layout: lanes 0-15 -> M=r, lanes 16-31 -> M=r+8)
#pragma unroll
  for (int j = 0; j < 4; ++j) {
    const int n = n0 + j * 16 + l16;
    const float bsc = bias[n];
#pragma unroll
    for (int i = 0; i < 2; ++i) {
#pragma unroll
      for (int r = 0; r < 8; ++r) {
        const int m = m0 + i * 16 + hv * 8 + r;
        C[(size_t)m * N + n] = acc[i][j][r] + bsc;
      }
    }
  }
}

// ---------------------------------------------------------------------------
// Softmax over `width` contiguous floats per (n,q,h) group, in place.
// ---------------------------------------------------------------------------
__global__ __launch_bounds__(256) void softmax_kernel(float* __restrict__ a,
                                                      int width, int total) {
  int t = blockIdx.x * blockDim.x + threadIdx.x;
  if (t >= total) return;
  float* p = a + (size_t)t * width;
  float m = p[0];
  for (int i = 1; i < width; ++i) m = fmaxf(m, p[i]);
  float s = 0.f;
  for (int i = 0; i < width; ++i) { float e = expf(p[i] - m); p[i] = e; s += e; }
  float inv = 1.0f / s;
  for (int i = 0; i < width; ++i) p[i] *= inv;
}

// ---------------------------------------------------------------------------
// Deformable bilinear sampling. One 32-lane group per (n,q,h); each lane
// owns 2 of the 64 head channels. All levels are 32x32, so the sampling
// coordinate simplifies to x = qx + off_x, y = qy + off_y.
// ---------------------------------------------------------------------------
__global__ __launch_bounds__(256) void sample_kernel(
    const float* __restrict__ value,   // rows (n*Lin + i), 1024 cols (h*64+d)
    const float* __restrict__ off,     // (N*Lq, NH*nL*NP*2)
    const float* __restrict__ attw,    // (N*Lq, NH*nL*NP) softmaxed
    float* __restrict__ out,           // (N*Lq, 1024)
    int nL, int Lin) {
  const int t = blockIdx.x * 256 + threadIdx.x;
  const int lane = t & 31;
  const int grp = t >> 5;              // (n*LQ+q)*NH + h
  const int h = grp & (NHn - 1);
  const int rq = grp >> 4;             // n*LQ + q
  const int q = rq & (LQn - 1);
  const int qx = q & (EE - 1);
  const int qy = q >> 5;
  const int n = rq >> 10;
  const int d0 = lane << 1;

  const float* vbase = value + ((size_t)n * Lin) * DDim + h * DHn + d0;
  const float* offp = off + (size_t)rq * (NHn * nL * NPn * 2) + h * (nL * NPn * 2);
  const float* awp = attw + (size_t)rq * (NHn * nL * NPn) + h * (nL * NPn);
  float acc0 = 0.f, acc1 = 0.f;
  for (int l = 0; l < nL; ++l) {
    const int lbase = l * (EE * EE);
    for (int p = 0; p < NPn; ++p) {
      const float ox = offp[(l * NPn + p) * 2 + 0];
      const float oy = offp[(l * NPn + p) * 2 + 1];
      const float aw = awp[l * NPn + p];
      const float x = (float)qx + ox;
      const float y = (float)qy + oy;
      const float x0f = floorf(x), y0f = floorf(y);
      const float fx = x - x0f, fy = y - y0f;
      const int x0 = (int)x0f, y0 = (int)y0f;
#pragma unroll
      for (int dy = 0; dy < 2; ++dy) {
        const int yi = y0 + dy;
        const float wy = dy ? fy : 1.f - fy;
#pragma unroll
        for (int dx = 0; dx < 2; ++dx) {
          const int xi = x0 + dx;
          const float wx = dx ? fx : 1.f - fx;
          if (xi >= 0 && xi < EE && yi >= 0 && yi < EE) {
            const float* vp = vbase + (size_t)(lbase + yi * EE + xi) * DDim;
            const float w = aw * wx * wy;
            acc0 += w * vp[0];
            acc1 += w * vp[1];
          }
        }
      }
    }
  }
  float* op = out + (size_t)rq * DDim + h * DHn + d0;
  op[0] = acc0;
  op[1] = acc1;
}

// ---------------------------------------------------------------------------
// out = src3 + gamma1 * (attn_ca + gamma2 * attn2)
// ---------------------------------------------------------------------------
__global__ __launch_bounds__(256) void combine_kernel(
    const float* __restrict__ src3, const float* __restrict__ g1,
    const float* __restrict__ g2, const float* __restrict__ attn,
    const float* __restrict__ attn2, float* __restrict__ out, size_t nelem) {
  size_t i = blockIdx.x * (size_t)blockDim.x + threadIdx.x;
  size_t stride = (size_t)gridDim.x * blockDim.x;
  for (; i < nelem; i += stride) {
    int c = (int)(i & (DDim - 1));
    out[i] = src3[i] + g1[c] * (attn[i] + g2[c] * attn2[i]);
  }
}

// ---------------------------------------------------------------------------
extern "C" void kernel_launch(void* const* d_in, const int* in_sizes, int n_in,
                              void* d_out, int out_size, void* d_ws, size_t ws_size,
                              hipStream_t stream) {
  (void)in_sizes; (void)n_in; (void)out_size; (void)ws_size;
  const float* src[4] = {(const float*)d_in[0], (const float*)d_in[1],
                         (const float*)d_in[2], (const float*)d_in[3]};
  const float* qn_g = (const float*)d_in[4];
  const float* qn_b = (const float*)d_in[5];
  const float* fn_g = (const float*)d_in[6];
  const float* fn_b = (const float*)d_in[7];
  const float* n1_g = (const float*)d_in[8];
  const float* n1_b = (const float*)d_in[9];
  const float* gamma1 = (const float*)d_in[10];
  const float* gamma2 = (const float*)d_in[11];
  const float* ca_vw = (const float*)d_in[12];
  const float* ca_vb = (const float*)d_in[13];
  const float* ca_ow = (const float*)d_in[14];
  const float* ca_ob = (const float*)d_in[15];
  const float* ca_aw = (const float*)d_in[16];
  const float* ca_ab = (const float*)d_in[17];
  const float* ca_pw = (const float*)d_in[18];
  const float* ca_pb = (const float*)d_in[19];
  const float* sa_vw = (const float*)d_in[20];
  const float* sa_vb = (const float*)d_in[21];
  const float* sa_ow = (const float*)d_in[22];
  const float* sa_ob = (const float*)d_in[23];
  const float* sa_aw = (const float*)d_in[24];
  const float* sa_ab = (const float*)d_in[25];
  const float* sa_pw = (const float*)d_in[26];
  const float* sa_pb = (const float*)d_in[27];
  float* out = (float*)d_out;

  // ---- carve workspace -----------------------------------------------------
  char* wsp = (char*)d_ws;
  size_t off = 0;
  auto alloc = [&](size_t bytes) -> void* {
    void* p = wsp + off;
    off = (off + bytes + 255) & ~(size_t)255;
    return p;
  };
  __bf16* feat_bf   = (__bf16*)alloc((size_t)16384 * 1024 * 2);
  __bf16* q_bf      = (__bf16*)alloc((size_t)4096 * 1024 * 2);
  __bf16* ca_vw_bf  = (__bf16*)alloc((size_t)1024 * 1024 * 2);
  __bf16* ca_ow_bf  = (__bf16*)alloc((size_t)512 * 1024 * 2);
  __bf16* ca_aw_bf  = (__bf16*)alloc((size_t)256 * 1024 * 2);
  __bf16* ca_pw_bf  = (__bf16*)alloc((size_t)1024 * 1024 * 2);
  __bf16* sa_vw_bf  = (__bf16*)alloc((size_t)1024 * 1024 * 2);
  __bf16* sa_ow_bf  = (__bf16*)alloc((size_t)128 * 1024 * 2);
  __bf16* sa_aw_bf  = (__bf16*)alloc((size_t)64 * 1024 * 2);
  __bf16* sa_pw_bf  = (__bf16*)alloc((size_t)1024 * 1024 * 2);
  float* value_ca   = (float*)alloc((size_t)16384 * 1024 * 4);
  float* off_ca     = (float*)alloc((size_t)4096 * 512 * 4);
  float* attw_ca    = (float*)alloc((size_t)4096 * 256 * 4);
  float* sampled    = (float*)alloc((size_t)4096 * 1024 * 4);
  __bf16* samp_bf   = (__bf16*)alloc((size_t)4096 * 1024 * 2);
  float* attn       = (float*)alloc((size_t)4096 * 1024 * 4);
  __bf16* attn1_bf  = (__bf16*)alloc((size_t)4096 * 1024 * 2);
  float* value_sa   = (float*)alloc((size_t)4096 * 1024 * 4);
  float* off_sa     = (float*)alloc((size_t)4096 * 128 * 4);
  float* attw_sa    = (float*)alloc((size_t)4096 * 64 * 4);
  float* sampled2   = (float*)alloc((size_t)4096 * 1024 * 4);
  float* attn2      = (float*)alloc((size_t)4096 * 1024 * 4);

  auto cvt = [&](const float* x, __bf16* y, size_t nelem) {
    size_t nvec = nelem / 4;
    int blocks = (int)((nvec + 255) / 256);
    if (blocks > 4096) blocks = 4096;
    cvt_bf16_kernel<<<blocks, 256, 0, stream>>>(x, y, nvec);
  };
  auto gemm = [&](const __bf16* A, const __bf16* W, const float* bias, float* C,
                  int M, int N) {
    dim3 grid(M / 256, N / 64);
    gemm_bf16_wmma<<<grid, 256, 0, stream>>>(A, W, bias, C, M, N);
  };

  // ---- weight conversion ---------------------------------------------------
  cvt(ca_vw, ca_vw_bf, (size_t)1024 * 1024);
  cvt(ca_ow, ca_ow_bf, (size_t)512 * 1024);
  cvt(ca_aw, ca_aw_bf, (size_t)256 * 1024);
  cvt(ca_pw, ca_pw_bf, (size_t)1024 * 1024);
  cvt(sa_vw, sa_vw_bf, (size_t)1024 * 1024);
  cvt(sa_ow, sa_ow_bf, (size_t)128 * 1024);
  cvt(sa_aw, sa_aw_bf, (size_t)64 * 1024);
  cvt(sa_pw, sa_pw_bf, (size_t)1024 * 1024);

  // ---- layernorms ----------------------------------------------------------
  for (int i = 0; i < 4; ++i)   // feat = concat(src0..3) rows: n*4096 + i*1024 + q
    ln_bf16_kernel<<<4096, 256, 0, stream>>>(src[i], fn_g, fn_b,
                                             feat_bf + (size_t)i * 1024 * 1024,
                                             1024, 4096);
  ln_bf16_kernel<<<4096, 256, 0, stream>>>(src[3], qn_g, qn_b, q_bf, 1024, 1024);

  // ---- cross attention (4 levels) -----------------------------------------
  gemm(feat_bf, ca_vw_bf, ca_vb, value_ca, 16384, 1024);
  gemm(q_bf, ca_ow_bf, ca_ob, off_ca, 4096, 512);
  gemm(q_bf, ca_aw_bf, ca_ab, attw_ca, 4096, 256);
  softmax_kernel<<<(65536 + 255) / 256, 256, 0, stream>>>(attw_ca, 16, 65536);
  sample_kernel<<<8192, 256, 0, stream>>>(value_ca, off_ca, attw_ca, sampled, 4, 4096);
  cvt(sampled, samp_bf, (size_t)4096 * 1024);
  gemm(samp_bf, ca_pw_bf, ca_pb, attn, 4096, 1024);

  // ---- self attention (1 level) -------------------------------------------
  ln_bf16_kernel<<<4096, 256, 0, stream>>>(attn, n1_g, n1_b, attn1_bf, 1024, 1024);
  gemm(attn1_bf, sa_vw_bf, sa_vb, value_sa, 4096, 1024);
  gemm(attn1_bf, sa_ow_bf, sa_ob, off_sa, 4096, 128);
  gemm(attn1_bf, sa_aw_bf, sa_ab, attw_sa, 4096, 64);
  softmax_kernel<<<(65536 + 255) / 256, 256, 0, stream>>>(attw_sa, 4, 65536);
  sample_kernel<<<8192, 256, 0, stream>>>(value_sa, off_sa, attw_sa, sampled2, 1, 1024);
  cvt(sampled2, samp_bf, (size_t)4096 * 1024);
  gemm(samp_bf, sa_pw_bf, sa_pb, attn2, 4096, 1024);

  // ---- residual combine ----------------------------------------------------
  combine_kernel<<<4096, 256, 0, stream>>>(src[3], gamma1, gamma2, attn, attn2,
                                           out, (size_t)4096 * 1024);
}
